// Graph_GAN_9414568313604
// MI455X (gfx1250) — matrix operand order repaired and checked by reading, not compiled, not run on hardware
//
#include <hip/hip_runtime.h>

#define LRELU_ALPHA 0.2f

typedef __attribute__((ext_vector_type(16))) __bf16 v16bf;
typedef __attribute__((ext_vector_type(8)))  float  v8f;

// ---------------------------------------------------------------------------
// Fragment loaders.
//
// A-fragment (16x32 bf16, MxK): lane L<16 holds row L, elements 0..7 = K 0..7,
// elements 8..15 = K 16..23; lanes 16..31 hold row L-16 with K 8..15 / 24..31.
// -> two contiguous 16B chunks per lane from a row-major activation buffer.
//
// B-fragment (32x16 bf16, KxN): lane L holds column L%16; element s is
// K = 8h+s (s<8), 8+8h+s (s>=8), h = L/16.  Weights are pre-swizzled so a fragment is 512 contiguous
// bf16 with element p*16 + t = K (p/16)*16 + t, col p%16 -> two 16B reads per lane.
// ---------------------------------------------------------------------------
__device__ __forceinline__ v16bf load2(const __bf16* p0, const __bf16* p1) {
    union { v16bf v; uint4 q[2]; } u;
    u.q[0] = *(const uint4*)p0;
    u.q[1] = *(const uint4*)p1;
    return u.v;
}

// Hoist all A-fragments of a layer into registers (invariant across nt).
template<int KT>
__device__ __forceinline__ void load_a(v16bf (&a)[KT], const __bf16* __restrict__ in,
                                       int strideIn, int lane) {
    const int hf  = lane >> 4;
    const int row = lane & 15;
#pragma unroll
    for (int kt = 0; kt < KT; ++kt)
        a[kt] = load2(in + row * strideIn + kt * 32 + hf * 8,
                      in + row * strideIn + kt * 32 + 16 + hf * 8);
}

// One 16x16 output tile: C = sum_kt A[kt] * B[kt][nt]
template<int KT, int NT>
__device__ __forceinline__ v8f mm_nt(const v16bf (&a)[KT], const __bf16* __restrict__ Wl,
                                     int nt, int lane) {
    v8f c;
#pragma unroll
    for (int r = 0; r < 8; ++r) c[r] = 0.0f;
#pragma unroll
    for (int kt = 0; kt < KT; ++kt) {
        const __bf16* wp = Wl + (kt * NT + nt) * 512 + (lane & 15) * 16 + (lane >> 4) * 8;
        v16bf bm = load2(wp, wp + 256);
        c = __builtin_amdgcn_wmma_f32_16x16x32_bf16(false, a[kt], false, bm,
                                                    (short)0, c, false, false);
    }
    return c;
}

// ---------------------------------------------------------------------------
// Weight pre-swizzle: row-major f32 W[K][N] -> bf16 B-fragment-major layout.
// ---------------------------------------------------------------------------
__global__ void swizzle_kernel(const float* __restrict__ W, __bf16* __restrict__ dst,
                               int K, int N) {
    int tid = blockIdx.x * blockDim.x + threadIdx.x;
    int total = (K >> 5) * (N >> 4) * 512;
    if (tid >= total) return;
    int frag = tid >> 9, e = tid & 511;
    int nTiles = N >> 4;
    int kt = frag / nTiles, nt = frag - kt * nTiles;
    int lane = e >> 4, t = e & 15;
    int k = kt * 32 + (lane >> 4) * 16 + t;
    int n = nt * 16 + (lane & 15);
    dst[tid] = (__bf16)W[k * N + n];
}

// ---------------------------------------------------------------------------
// Edge kernel: one workgroup per node (b,i).  7 waves, each owns 16 j-rows.
// Fused: edge MLP (64->96->160->192, leaky) + sum over j.  Writes h-row
// [sum(192) | x_i(32)] as bf16 for the node MLP.
// LDS: 52224 bf16 weights + 7 * 4096 bf16 ping-pong activations + 192 f32 sum
//      = 162560 bytes.
// ---------------------------------------------------------------------------
__global__ void __launch_bounds__(224) edge_kernel(
        const float* __restrict__ x, const __bf16* __restrict__ wfe,
        const float* __restrict__ bias1, const float* __restrict__ bias2,
        const float* __restrict__ bias3, __bf16* __restrict__ hbuf) {
    extern __shared__ __align__(16) char smem[];
    __bf16* sW   = (__bf16*)smem;              // 52224 elems (layers packed)
    __bf16* sBuf = sW + 52224;                 // 7 waves * 4096 elems
    float*  sSum = (float*)(sBuf + 7 * 4096);  // 192 floats

    const int tid  = threadIdx.x;
    const int wave = tid >> 5, lane = tid & 31;
    const int hf   = lane >> 4, col = lane & 15;
    const int b = blockIdx.x / 100;
    const int i = blockIdx.x - b * 100;
    const float* xb = x + b * 3200;            // 100 rows * 32 feats

    // Stage 104448 B of swizzled weights straight into LDS with the gfx1250
    // async copy path (no VGPR round-trip, tracked by ASYNCcnt).
    {
        unsigned ldsBase = (unsigned)(size_t)sW;   // flat->LDS uses addr[31:0]
        for (int idx = tid; idx < 6528; idx += 224) {
            unsigned off = (unsigned)idx * 16u;
            unsigned lds = ldsBase + off;
            asm volatile("global_load_async_to_lds_b128 %0, %1, %2 offset:0"
                         :: "v"(lds), "v"(off), "s"(wfe) : "memory");
        }
        asm volatile("s_wait_asynccnt 0" ::: "memory");
    }
    if (tid < 192) sSum[tid] = 0.0f;

    __bf16* bufA = sBuf + wave * 4096;   // up to 16x160
    __bf16* bufB = bufA + 2560;          // 16x96
    const int j0 = wave * 16;

    // build layer-0 input tile [16 x 64] = [x_i | x_j], zero-pad j>=100
    for (int idx = lane; idx < 1024; idx += 32) {
        int r = idx >> 6, k = idx & 63;
        int j = j0 + r;
        float v = (k < 32) ? xb[i * 32 + k]
                           : ((j < 100) ? xb[j * 32 + (k - 32)] : 0.0f);
        bufA[idx] = (__bf16)v;
    }
    __syncthreads();   // weights + sSum ready

    // ---- layer 1: 64 -> 96 ----
    {
        v16bf a[2];
        load_a<2>(a, bufA, 64, lane);
        for (int nt = 0; nt < 6; ++nt) {
            v8f c = mm_nt<2, 6>(a, sW, nt, lane);
            float bv = bias1[nt * 16 + col];
#pragma unroll
            for (int r = 0; r < 8; ++r) {
                float v = c[r] + bv;
                v = v > 0.0f ? v : LRELU_ALPHA * v;
                bufB[(r + 8 * hf) * 96 + nt * 16 + col] = (__bf16)v;
            }
        }
    }
    // ---- layer 2: 96 -> 160 ----
    {
        v16bf a[3];
        load_a<3>(a, bufB, 96, lane);
        for (int nt = 0; nt < 10; ++nt) {
            v8f c = mm_nt<3, 10>(a, sW + 6144, nt, lane);
            float bv = bias2[nt * 16 + col];
#pragma unroll
            for (int r = 0; r < 8; ++r) {
                float v = c[r] + bv;
                v = v > 0.0f ? v : LRELU_ALPHA * v;
                bufA[(r + 8 * hf) * 160 + nt * 16 + col] = (__bf16)v;
            }
        }
    }
    // ---- layer 3: 160 -> 192, leaky, masked sum over j ----
    {
        v16bf a[5];
        load_a<5>(a, bufA, 160, lane);
        for (int nt = 0; nt < 12; ++nt) {
            v8f c = mm_nt<5, 12>(a, sW + 21504, nt, lane);
            float bv = bias3[nt * 16 + col];
            float s = 0.0f;
#pragma unroll
            for (int r = 0; r < 8; ++r) {
                float v = c[r] + bv;
                v = v > 0.0f ? v : LRELU_ALPHA * v;
                if (j0 + r + 8 * hf < 100) s += v;   // mask padded j rows
            }
            s += __shfl_xor(s, 16, 32);              // fold lane halves (same col)
            if (hf == 0) atomicAdd(&sSum[nt * 16 + col], s);   // ds_add_f32
        }
    }
    __syncthreads();

    // h row = [S(192) | x_i(32)] in bf16
    {
        long rowOff = (long)(b * 100 + i) * 224;
        if (tid < 192)      hbuf[rowOff + tid] = (__bf16)sSum[tid];
        else if (tid < 224) hbuf[rowOff + tid] = (__bf16)xb[i * 32 + (tid - 192)];
    }
}

// ---------------------------------------------------------------------------
// Node kernel: 6400 rows = 400 tiles of 16, 1 wave per tile, 8 waves/block.
// 224 -> 256 -> 256 -> 32 (leaky, leaky, linear).  B-fragments from L2.
// LDS: 8 waves * 2 * (16x256) bf16 = 131072 bytes.
// ---------------------------------------------------------------------------
__global__ void __launch_bounds__(256) node_kernel(
        const __bf16* __restrict__ hbuf, const __bf16* __restrict__ wfn,
        const float* __restrict__ bias1, const float* __restrict__ bias2,
        const float* __restrict__ bias3, float* __restrict__ xout) {
    extern __shared__ __align__(16) char smem[];
    const int tid  = threadIdx.x;
    const int wave = tid >> 5, lane = tid & 31;
    const int hf   = lane >> 4, col = lane & 15;
    __bf16* bufA = (__bf16*)smem + wave * 8192;   // 16x256
    __bf16* bufB = bufA + 4096;                   // 16x256
    const int tile = blockIdx.x * 8 + wave;       // 0..399
    const __bf16* in0 = hbuf + (long)tile * 16 * 224;

    // warm layer-2/3 weight fragments into the caches (global_prefetch_b8)
    __builtin_prefetch(wfn + 57344 + tid * 128, 0, 2);
    __builtin_prefetch(wfn + 122880 + tid * 32, 0, 2);

    // ---- layer 1: 224 -> 256 (A direct from global) ----
    {
        v16bf a[7];
        load_a<7>(a, in0, 224, lane);
        for (int nt = 0; nt < 16; ++nt) {
            v8f c = mm_nt<7, 16>(a, wfn, nt, lane);
            float bv = bias1[nt * 16 + col];
#pragma unroll
            for (int r = 0; r < 8; ++r) {
                float v = c[r] + bv;
                v = v > 0.0f ? v : LRELU_ALPHA * v;
                bufA[(r + 8 * hf) * 256 + nt * 16 + col] = (__bf16)v;
            }
        }
    }
    // ---- layer 2: 256 -> 256 ----
    {
        v16bf a[8];
        load_a<8>(a, bufA, 256, lane);
        for (int nt = 0; nt < 16; ++nt) {
            v8f c = mm_nt<8, 16>(a, wfn + 57344, nt, lane);
            float bv = bias2[nt * 16 + col];
#pragma unroll
            for (int r = 0; r < 8; ++r) {
                float v = c[r] + bv;
                v = v > 0.0f ? v : LRELU_ALPHA * v;
                bufB[(r + 8 * hf) * 256 + nt * 16 + col] = (__bf16)v;
            }
        }
    }
    // ---- layer 3: 256 -> 32, linear, f32 out ----
    {
        v16bf a[8];
        load_a<8>(a, bufB, 256, lane);
        for (int nt = 0; nt < 2; ++nt) {
            v8f c = mm_nt<8, 2>(a, wfn + 122880, nt, lane);
            float bv = bias3[nt * 16 + col];
#pragma unroll
            for (int r = 0; r < 8; ++r) {
                int m = r + 8 * hf;
                xout[((long)tile * 16 + m) * 32 + nt * 16 + col] = c[r] + bv;
            }
        }
    }
}

__global__ void tanh_kernel(const float* __restrict__ xf, float* __restrict__ out) {
    int idx = blockIdx.x * blockDim.x + threadIdx.x;
    if (idx < 64 * 100 * 3) {
        int row = idx / 3, c = idx - row * 3;
        out[idx] = tanhf(xf[row * 32 + c]);
    }
}

// ---------------------------------------------------------------------------
// Launch
// ---------------------------------------------------------------------------
extern "C" void kernel_launch(void* const* d_in, const int* in_sizes, int n_in,
                              void* d_out, int out_size, void* d_ws, size_t ws_size,
                              hipStream_t stream) {
    (void)in_sizes; (void)n_in; (void)out_size; (void)ws_size;

    const float* x0 = (const float*)d_in[0];

    // workspace layout (bytes)
    char* ws = (char*)d_ws;
    __bf16* wsz  = (__bf16*)ws;                    // swizzled weights: 733184 B
    __bf16* hbuf = (__bf16*)(ws + 733184);         // 6400 x 224 bf16: 2867200 B
    float*  x1   = (float*)(ws + 3600384);         // 6400 x 32 f32
    float*  x2   = (float*)(ws + 4419584);         // 6400 x 32 f32

    // element offsets inside the swizzled-weight pool
    const int feOff[3] = {0, 6144, 21504};         // fe layer bases (per iter: 52224)
    const int fnOff[3] = {0, 57344, 122880};       // fn layer bases (per iter: 131072)
    const int feK[3] = {64, 96, 160},  feN[3] = {96, 160, 192};
    const int fnK[3] = {224, 256, 256}, fnN[3] = {256, 256, 32};

    for (int it = 0; it < 2; ++it)
        for (int l = 0; l < 3; ++l) {
            const float* W = (const float*)d_in[1 + it * 6 + l * 2];
            int total = (feK[l] >> 5) * (feN[l] >> 4) * 512;
            swizzle_kernel<<<(total + 255) / 256, 256, 0, stream>>>(
                W, wsz + it * 52224 + feOff[l], feK[l], feN[l]);
        }
    for (int it = 0; it < 2; ++it)
        for (int l = 0; l < 3; ++l) {
            const float* W = (const float*)d_in[13 + it * 6 + l * 2];
            int total = (fnK[l] >> 5) * (fnN[l] >> 4) * 512;
            swizzle_kernel<<<(total + 255) / 256, 256, 0, stream>>>(
                W, wsz + 104448 + it * 131072 + fnOff[l], fnK[l], fnN[l]);
        }

    const int EDGE_SMEM = 162560;
    const int NODE_SMEM = 131072;
    (void)hipFuncSetAttribute((const void*)edge_kernel,
                              hipFuncAttributeMaxDynamicSharedMemorySize, EDGE_SMEM);
    (void)hipFuncSetAttribute((const void*)node_kernel,
                              hipFuncAttributeMaxDynamicSharedMemorySize, NODE_SMEM);

    const float* xin = x0;
    float* xouts[2] = {x1, x2};
    for (int it = 0; it < 2; ++it) {
        const float* fb1 = (const float*)d_in[2 + it * 6 + 0];
        const float* fb2 = (const float*)d_in[2 + it * 6 + 2];
        const float* fb3 = (const float*)d_in[2 + it * 6 + 4];
        edge_kernel<<<6400, 224, EDGE_SMEM, stream>>>(
            xin, wsz + it * 52224, fb1, fb2, fb3, hbuf);

        const float* nb1 = (const float*)d_in[14 + it * 6 + 0];
        const float* nb2 = (const float*)d_in[14 + it * 6 + 2];
        const float* nb3 = (const float*)d_in[14 + it * 6 + 4];
        node_kernel<<<50, 256, NODE_SMEM, stream>>>(
            hbuf, wsz + 104448 + it * 131072, nb1, nb2, nb3, xouts[it]);
        xin = xouts[it];
    }

    tanh_kernel<<<(19200 + 255) / 256, 256, 0, stream>>>(x2, (float*)d_out);
}
